// GDNModel_57621281243709
// MI455X (gfx1250) — compile-verified, matrix operand order checked by reading, not run on hardware
//
#include <hip/hip_runtime.h>
#include <math.h>

// Problem constants (from reference)
#define BB 64
#define NN 1024
#define F_IN 64
#define EE 128
#define TOPK 32
#define NEG_SLOPE 0.2f
#define M_TOTAL (BB * NN)   // 65536 rows of h

typedef __attribute__((ext_vector_type(16))) _Float16 v16h;
typedef __attribute__((ext_vector_type(8)))  float    v8f;
typedef __attribute__((ext_vector_type(4)))  float    v4f;

// ---------------------------------------------------------------------------
// Kernel 0: reciprocal norms of emb rows.  rnorm[j] = 1/||emb[j]||
// ---------------------------------------------------------------------------
__global__ void rnorm_kernel(const float* __restrict__ emb,
                             float* __restrict__ rnorm) {
  int j = blockIdx.x * blockDim.x + threadIdx.x;   // 0..1023
  if (j >= NN) return;
  const v4f* er = (const v4f*)(emb + (size_t)j * EE);
  float s = 0.f;
#pragma unroll
  for (int e = 0; e < EE / 4; ++e) {
    v4f v = er[e];
    s += v[0] * v[0] + v[1] * v[1] + v[2] * v[2] + v[3] * v[3];
  }
  rnorm[j] = rsqrtf(s);
}

// ---------------------------------------------------------------------------
// Kernel 1: per-row cosine similarity + top-32 indices (lowest-index ties,
// matching jax.lax.top_k).  One block (128 threads) per emb row.
// ---------------------------------------------------------------------------
__global__ void topk_kernel(const float* __restrict__ emb,
                            const float* __restrict__ rnorm,
                            int* __restrict__ idx) {
  __shared__ float sims[NN];
  __shared__ float srow[EE];
  __shared__ float bval[128];
  __shared__ int   bidx[128];

  int i = blockIdx.x;
  int tid = threadIdx.x;   // 0..127

  srow[tid] = emb[(size_t)i * EE + tid];
  __syncthreads();

  float rni = rnorm[i];
  for (int j = tid; j < NN; j += 128) {
    const v4f* ej = (const v4f*)(emb + (size_t)j * EE);
    float dot = 0.f;
#pragma unroll 8
    for (int e = 0; e < EE / 4; ++e) {
      v4f v = ej[e];
      dot += srow[4 * e] * v[0] + srow[4 * e + 1] * v[1] +
             srow[4 * e + 2] * v[2] + srow[4 * e + 3] * v[3];
    }
    sims[j] = dot * rni * rnorm[j];
  }
  __syncthreads();

  for (int k = 0; k < TOPK; ++k) {
    float bv = -INFINITY;
    int   bi = 0x7fffffff;
    for (int j = tid; j < NN; j += 128) {
      float v = sims[j];
      if (v > bv || (v == bv && j < bi)) { bv = v; bi = j; }
    }
    bval[tid] = bv;
    bidx[tid] = bi;
    __syncthreads();
    for (int s = 64; s > 0; s >>= 1) {
      if (tid < s) {
        float v2 = bval[tid + s];
        int   i2 = bidx[tid + s];
        if (v2 > bval[tid] || (v2 == bval[tid] && i2 < bidx[tid])) {
          bval[tid] = v2; bidx[tid] = i2;
        }
      }
      __syncthreads();
    }
    if (tid == 0) {
      idx[i * TOPK + k] = bidx[0];
      sims[bidx[0]] = -INFINITY;
    }
    __syncthreads();
  }
}

// ---------------------------------------------------------------------------
// Kernel 2: h = x @ W_fe  via v_wmma_f32_16x16x32_f16.
// M = 65536, K = 64 (two K=32 WMMA accumulations), Ncols = 128 (8 n-tiles).
// Block = 256 threads = 8 waves; each wave owns one 16-row m-tile and sweeps
// all 8 n-tiles, reusing its A fragments.
//
// W_fe is staged into LDS PRE-SWIZZLED into WMMA B-fragment layout: one v16h
// per (ntile, kfrag, lane), so the inner loop does a single 32-byte ds load
// (2x ds_load_b128) per fragment instead of 32x ds_load_u16.
// A fragments come from 4 aligned float4 global loads per fragment.
// ---------------------------------------------------------------------------
__global__ void gemm_h_kernel(const float* __restrict__ x,
                              const float* __restrict__ Wfe,
                              float* __restrict__ h) {
  // [ntile][kfrag][lane] -> 16 halves (32B).  8*2*32 = 512 frags = 16 KB.
  __shared__ v16h sWv[8 * 2 * 32];
  _Float16* sW = (_Float16*)sWv;

  int tid = threadIdx.x;
  // Stage + swizzle: source-linear walk (coalesced global reads).
  // src idx = k*128 + n.  Fragment coords:
  //   kk = k & 31, f = k >> 5
  //   laneHi = (kk >> 3) & 1      (lanes 16-31 hold kk in {8..15, 24..31})
  //   i      = (kk & 7) + ((kk >> 4) & 1) * 8
  //   lane   = laneHi*16 + (n & 15),  nt = n >> 4
  for (int s = tid; s < F_IN * EE; s += 256) {
    int k = s >> 7;
    int n = s & 127;
    int kk = k & 31;
    int f  = k >> 5;
    int laneHi = (kk >> 3) & 1;
    int i  = (kk & 7) + ((kk >> 4) & 1) * 8;
    int lane = laneHi * 16 + (n & 15);
    int nt = n >> 4;
    sW[(((nt * 2 + f) * 32) + lane) * 16 + i] = (_Float16)Wfe[s];
  }
  __syncthreads();

  int wave = tid >> 5;
  int lane = tid & 31;
  int mtile = blockIdx.x * 8 + wave;          // 0..4095
  int mrow  = (mtile << 4) + (lane & 15);
  int khalf = (lane >> 4) * 8;                // lanes 16-31 hold K+8 halves
  const v4f* xrow = (const v4f*)(x + (size_t)mrow * F_IN);

  // A fragments: elem i -> K = khalf + (i<8 ? i : i+8).  The 16 halves are
  // two contiguous 8-float runs -> 4 float4 loads per fragment.
  int q = khalf >> 2;                         // 0 or 2
  v4f p0 = xrow[q + 0], p1 = xrow[q + 1];     // K = khalf..khalf+7
  v4f p2 = xrow[q + 4], p3 = xrow[q + 5];     // K = khalf+16..khalf+23
  v4f p4 = xrow[q + 8], p5 = xrow[q + 9];     // +32
  v4f p6 = xrow[q + 12], p7 = xrow[q + 13];
  v16h a0, a1;
#pragma unroll
  for (int i = 0; i < 4; ++i) {
    a0[i]      = (_Float16)p0[i];
    a0[i + 4]  = (_Float16)p1[i];
    a0[i + 8]  = (_Float16)p2[i];
    a0[i + 12] = (_Float16)p3[i];
    a1[i]      = (_Float16)p4[i];
    a1[i + 4]  = (_Float16)p5[i];
    a1[i + 8]  = (_Float16)p6[i];
    a1[i + 12] = (_Float16)p7[i];
  }

  int mbase = (mtile << 4) + (lane >> 4) * 8;
#pragma unroll 1
  for (int nt = 0; nt < 8; ++nt) {
    int ncol = nt * 16 + (lane & 15);
    v16h b0 = sWv[(nt * 2 + 0) * 32 + lane];   // 2x ds_load_b128
    v16h b1 = sWv[(nt * 2 + 1) * 32 + lane];
    v8f c = {};
    c = __builtin_amdgcn_wmma_f32_16x16x32_f16(false, a0, false, b0,
                                               (short)0, c, false, false);
    c = __builtin_amdgcn_wmma_f32_16x16x32_f16(false, a1, false, b1,
                                               (short)0, c, false, false);
    // D layout: lane l, vgpr r -> (m = mbase + r, n = ncol)
#pragma unroll
    for (int r = 0; r < 8; ++r)
      h[(size_t)(mbase + r) * EE + ncol] = c[r];
  }
}

// ---------------------------------------------------------------------------
// Kernel 3: attention logits e_s, e_d.  One thread per (b,n).
// e_s[m] = h[m]·a_src[:128] + emb[n]·a_src[128:], same for e_d with a_dst.
// ---------------------------------------------------------------------------
__global__ void logits_kernel(const float* __restrict__ h,
                              const float* __restrict__ emb,
                              const float* __restrict__ a_src,
                              const float* __restrict__ a_dst,
                              float* __restrict__ es,
                              float* __restrict__ ed) {
  __shared__ float sa[2 * EE];
  __shared__ float sd[2 * EE];
  int tid = threadIdx.x;           // 0..255
  sa[tid] = a_src[tid];
  sd[tid] = a_dst[tid];
  __syncthreads();

  int m = blockIdx.x * 256 + tid;  // 0..65535
  int n = m & (NN - 1);
  const v4f* hr = (const v4f*)(h + (size_t)m * EE);
  const v4f* er = (const v4f*)(emb + (size_t)n * EE);
  float s = 0.f, d = 0.f;
#pragma unroll 4
  for (int e = 0; e < EE / 4; ++e) {
    v4f hv = hr[e], ev = er[e];
#pragma unroll
    for (int c = 0; c < 4; ++c) {
      s += hv[c] * sa[4 * e + c] + ev[c] * sa[EE + 4 * e + c];
      d += hv[c] * sd[4 * e + c] + ev[c] * sd[EE + 4 * e + c];
    }
  }
  es[m] = s;
  ed[m] = d;
}

// ---------------------------------------------------------------------------
// Kernel 4: per-(b,n) softmax over 32 neighbors + weighted gather-reduce of
// h rows + ReLU + final 128->1 projection.  One wave32 per (b,n); lane k owns
// neighbor k.  The h gather hits L2 (h = 32 MB << 192 MB L2).
// ---------------------------------------------------------------------------
__global__ void attn_out_kernel(const float* __restrict__ h,
                                const int* __restrict__ idx,
                                const float* __restrict__ es,
                                const float* __restrict__ ed,
                                const float* __restrict__ lin_W,
                                const float* __restrict__ lin_b,
                                float* __restrict__ y) {
  int wave = threadIdx.x >> 5;
  int lane = threadIdx.x & 31;
  int m = blockIdx.x * 8 + wave;           // 0..65535
  int b = m >> 10;
  int n = m & (NN - 1);

  int j = idx[n * TOPK + lane];            // neighbor index for k = lane
  float s = es[m] + ed[b * NN + j];
  s = (s >= 0.f) ? s : NEG_SLOPE * s;      // leaky relu

  // softmax across the 32 lanes
  float mx = s;
#pragma unroll
  for (int o = 16; o > 0; o >>= 1) mx = fmaxf(mx, __shfl_xor(mx, o, 32));
  float p = expf(s - mx);
  float sum = p;
#pragma unroll
  for (int o = 16; o > 0; o >>= 1) sum += __shfl_xor(sum, o, 32);
  float alpha = p / sum;

  // out[e] = sum_k alpha_k * h[b, idx_k, e]; lane covers e = c*32 + lane
  float yacc = 0.f;
#pragma unroll 1
  for (int c = 0; c < 4; ++c) {
    int e = c * 32 + lane;
    float acc = 0.f;
#pragma unroll
    for (int k = 0; k < TOPK; ++k) {
      float ak = __shfl(alpha, k, 32);
      int   jk = __shfl(j, k, 32);
      acc += ak * h[((size_t)b * NN + jk) * EE + e];
    }
    acc = (acc > 0.f) ? acc : 0.f;         // relu
    yacc += acc * lin_W[e];
  }
#pragma unroll
  for (int o = 16; o > 0; o >>= 1) yacc += __shfl_xor(yacc, o, 32);
  if (lane == 0) y[m] = yacc + lin_b[0];
}

// ---------------------------------------------------------------------------
// Host-side launcher
// ---------------------------------------------------------------------------
extern "C" void kernel_launch(void* const* d_in, const int* in_sizes, int n_in,
                              void* d_out, int out_size, void* d_ws, size_t ws_size,
                              hipStream_t stream) {
  const float* x     = (const float*)d_in[0];   // (64,1024,64)
  const float* emb   = (const float*)d_in[1];   // (1024,128)
  const float* Wfe   = (const float*)d_in[2];   // (64,128)
  const float* a_src = (const float*)d_in[3];   // (256,)
  const float* a_dst = (const float*)d_in[4];   // (256,)
  const float* lin_W = (const float*)d_in[5];   // (128,1)
  const float* lin_b = (const float*)d_in[6];   // (1,)
  float* y = (float*)d_out;                     // (64,1024)

  // Workspace layout
  char* ws = (char*)d_ws;
  float* h     = (float*)ws;                                  // 32 MB
  size_t off   = (size_t)M_TOTAL * EE * sizeof(float);
  float* rnorm = (float*)(ws + off);   off += NN * sizeof(float);
  int*   idx   = (int*)(ws + off);     off += (size_t)NN * TOPK * sizeof(int);
  float* es    = (float*)(ws + off);   off += (size_t)M_TOTAL * sizeof(float);
  float* ed    = (float*)(ws + off);   off += (size_t)M_TOTAL * sizeof(float);
  (void)ws_size; (void)in_sizes; (void)n_in; (void)out_size;

  rnorm_kernel<<<NN / 256, 256, 0, stream>>>(emb, rnorm);
  topk_kernel<<<NN, 128, 0, stream>>>(emb, rnorm, idx);
  gemm_h_kernel<<<M_TOTAL / 16 / 8, 256, 0, stream>>>(x, Wfe, h);
  logits_kernel<<<M_TOTAL / 256, 256, 0, stream>>>(h, emb, a_src, a_dst, es, ed);
  attn_out_kernel<<<M_TOTAL / 8, 256, 0, stream>>>(h, idx, es, ed, lin_W, lin_b, y);
}
